// TransformerBlock_5282809774085
// MI455X (gfx1250) — matrix operand order repaired and checked by reading, not compiled, not run on hardware
//
#include <hip/hip_runtime.h>

#define DEV static __device__ __forceinline__

typedef __attribute__((ext_vector_type(16))) __bf16 v16bf;
typedef __attribute__((ext_vector_type(8)))  __bf16 v8bf;
typedef __attribute__((ext_vector_type(8)))  float  v8f;

// Problem sizes (fixed by the reference)
constexpr int Bq   = 4;
constexpr int Nseq = 2048;
constexpr int Dm   = 1024;
constexpr int Hh   = 16;
constexpr int dhd  = 64;
constexpr int Ff   = 4096;
constexpr int BN   = Bq * Nseq;                     // 8192 token rows
constexpr size_t QS = (size_t)Bq * Hh * Nseq * dhd; // elems per q/k/v buffer

DEV __bf16 f2bf(float f) {
  unsigned u = __builtin_bit_cast(unsigned, f);
  unsigned r = (u + 0x7FFFu + ((u >> 16) & 1u)) >> 16;  // RNE
  unsigned short s = (unsigned short)r;
  return __builtin_bit_cast(__bf16, s);
}

DEV v16bf mk16(v8bf lo, v8bf hi) {
  return __builtin_shufflevector(lo, hi, 0,1,2,3,4,5,6,7,8,9,10,11,12,13,14,15);
}

// CDNA5 async global->LDS copy, 16B per lane. Tracked by ASYNCcnt; the LDS
// address VGPR is the low 32 bits of the flat shared pointer (flat->group
// addrspacecast is a truncate on AMDGPU).
DEV void async_copy_b128(void* lds_dst, const void* gsrc) {
  unsigned l = (unsigned)(uintptr_t)lds_dst;
  asm volatile("global_load_async_to_lds_b128 %0, %1, off"
               :: "v"(l), "v"(gsrc) : "memory");
}
DEV void wait_async0() { asm volatile("s_wait_asynccnt 0" ::: "memory"); }

// ---------------------------------------------------------------- convert
__global__ __launch_bounds__(256) void cvt_bf16_kernel(
    const float* __restrict__ src, __bf16* __restrict__ dst, size_t n) {
  size_t i = (size_t)blockIdx.x * blockDim.x + threadIdx.x;
  size_t stride = (size_t)gridDim.x * blockDim.x;
  for (; i < n; i += stride) dst[i] = f2bf(src[i]);
}

// ---------------------------------------------------------------- GEMM
// C[M,Nc] = A[M,K](bf16) @ Bw[K,Nc](bf16) + bias, f32 accumulation via
// v_wmma_f32_16x16x32_bf16.  Block tile 128x128, 8 waves (2x4), wave tile
// 64x32 (4x2 WMMA tiles).  A tile staged with global_load_async_to_lds_b128.
// EPI: 0 = scatter q/k/v -> [B,H,N,dh] bf16, 1 = exact GELU -> bf16,
//      2 = plain f32 row-major.
template <int EPI>
__global__ __launch_bounds__(256) void gemm_bf16_kernel(
    const __bf16* __restrict__ A, const __bf16* __restrict__ Bw,
    const float* __restrict__ bias, int K, int Nc,
    __bf16* __restrict__ outB, float* __restrict__ outF) {
  __shared__ __bf16 As[128 * 32];  // [m][k] row-major
  __shared__ __bf16 Bs[128 * 32];  // [n][k] (K-transposed for contiguous frags)

  const int tid  = threadIdx.x;
  const int lane = tid & 31, wid = tid >> 5;
  const int waveM = wid & 1, waveN = wid >> 1;   // 2 x 4 wave grid
  const int bm = blockIdx.y, bn = blockIdx.x;

  const v8f zero = {0.f, 0.f, 0.f, 0.f, 0.f, 0.f, 0.f, 0.f};
  v8f acc[4][2];
#pragma unroll
  for (int mt = 0; mt < 4; ++mt)
#pragma unroll
    for (int nt = 0; nt < 2; ++nt) acc[mt][nt] = zero;

  const int arow = tid >> 1, akh = (tid & 1) * 16;   // A loader: 2 thr/row
  const int bkr  = tid >> 3, bch = (tid & 7) * 16;   // B loader: 8 thr/row

  for (int k0 = 0; k0 < K; k0 += 32) {
    {  // A tile 128x32: async straight into LDS (no VGPR round trip)
      const __bf16* ga = A + (size_t)(bm * 128 + arow) * K + k0 + akh;
      __bf16* la = &As[arow * 32 + akh];
      async_copy_b128(la, ga);
      async_copy_b128(la + 8, ga + 8);
    }
    {  // B tile 32x128 -> LDS [n][k] (needs K-transpose, so via registers)
      const __bf16* gb = Bw + (size_t)(k0 + bkr) * Nc + bn * 128 + bch;
      alignas(16) __bf16 tmp[16];
      *(uint4*)&tmp[0] = *(const uint4*)gb;
      *(uint4*)&tmp[8] = *(const uint4*)(gb + 8);
#pragma unroll
      for (int i = 0; i < 16; ++i) Bs[(bch + i) * 32 + bkr] = tmp[i];
    }
    if (k0 + 32 < K) {  // next tiles -> global_prefetch_b8 (stays in L2)
      __builtin_prefetch(A + (size_t)(bm * 128 + arow) * K + k0 + 32 + akh, 0, 1);
      __builtin_prefetch(Bw + (size_t)(k0 + 32 + bkr) * Nc + bn * 128 + bch, 0, 1);
    }
    wait_async0();
    __syncthreads();

    // A frag: lane=row, 16 contiguous K at (l>>4)*16
    v16bf afrag[4], bfrag[2];
    const int kb = (lane >> 4) * 16;
#pragma unroll
    for (int mt = 0; mt < 4; ++mt) {
      const __bf16* p = &As[(waveM * 64 + mt * 16 + (lane & 15)) * 32 + kb];
      afrag[mt] = mk16(*(const v8bf*)p, *(const v8bf*)(p + 8));
    }
    // B frag: lane=col, 16 contiguous K at (l>>4)*16
    const int kb2 = (lane >> 4) * 16;
#pragma unroll
    for (int nt = 0; nt < 2; ++nt) {
      const __bf16* p = &Bs[(waveN * 32 + nt * 16 + (lane & 15)) * 32 + kb2];
      bfrag[nt] = mk16(*(const v8bf*)p, *(const v8bf*)(p + 8));
    }
#pragma unroll
    for (int mt = 0; mt < 4; ++mt)
#pragma unroll
      for (int nt = 0; nt < 2; ++nt)
        acc[mt][nt] = __builtin_amdgcn_wmma_f32_16x16x32_bf16(
            false, afrag[mt], false, bfrag[nt], (short)0, acc[mt][nt], false, false);
    __syncthreads();
  }

  // Epilogue.  C layout: row = (lane>>4)*8 + r, col = lane&15 within tile.
#pragma unroll
  for (int mt = 0; mt < 4; ++mt) {
#pragma unroll
    for (int nt = 0; nt < 2; ++nt) {
      const int col = bn * 128 + waveN * 32 + nt * 16 + (lane & 15);
      const float bv = bias[col];
#pragma unroll
      for (int r = 0; r < 8; ++r) {
        const int row = bm * 128 + waveM * 64 + mt * 16 + (lane >> 4) * 8 + r;
        float v = acc[mt][nt][r] + bv;
        if (EPI == 0) {
          int which = col >> 10, d = col & 1023;
          int h = d >> 6, dc = d & 63;
          int b = row >> 11, n = row & 2047;
          outB[(size_t)which * QS + (((size_t)b * Hh + h) * Nseq + n) * dhd + dc] = f2bf(v);
        } else if (EPI == 1) {
          float gl = 0.5f * v * (1.0f + erff(v * 0.70710678118654752f));
          outB[(size_t)row * Nc + col] = f2bf(gl);
        } else {
          outF[(size_t)row * Nc + col] = v;
        }
      }
    }
  }
}

// ---------------------------------------------------------------- attention
// Flash-style: block = 128 query rows of one (b,h); wave owns 16 rows.
// Per 32-key tile: 4 WMMAs for S=QK^T, online softmax, P staged through
// per-wave LDS (C-layout -> A-layout), 4 WMMAs for O += P·V.
__global__ __launch_bounds__(256) void attn_kernel(
    const __bf16* __restrict__ qkv, float* __restrict__ out) {
  __shared__ __bf16 Ks[32 * 64];        // [kv][dh]
  __shared__ __bf16 Vst[64 * 32];       // [dh][kv] (transposed)
  __shared__ __bf16 Ps[8 * 16 * 32];    // per-wave P staging [16][32]

  const int tid = threadIdx.x, lane = tid & 31, wid = tid >> 5;
  const int bh = blockIdx.y, b = bh >> 4, h = bh & 15;
  const int qr0 = blockIdx.x * 128 + wid * 16;

  const __bf16* qb = qkv +          (size_t)bh * Nseq * dhd;
  const __bf16* kb = qkv + QS     + (size_t)bh * Nseq * dhd;
  const __bf16* vb = qkv + 2 * QS + (size_t)bh * Nseq * dhd;

  // Q fragments: two A-frags covering dh halves (K=0..31, 32..63)
  v16bf qf[2];
#pragma unroll
  for (int c = 0; c < 2; ++c) {
    const __bf16* p = qb + (size_t)(qr0 + (lane & 15)) * dhd + c * 32 + (lane >> 4) * 16;
    qf[c] = mk16(*(const v8bf*)p, *(const v8bf*)(p + 8));
  }

  const v8f zero = {0.f, 0.f, 0.f, 0.f, 0.f, 0.f, 0.f, 0.f};
  v8f acc[4];
#pragma unroll
  for (int o = 0; o < 4; ++o) acc[o] = zero;
  float m_run[8], l_run[8];
#pragma unroll
  for (int r = 0; r < 8; ++r) { m_run[r] = -1e30f; l_run[r] = 0.f; }

  const int lrow = tid >> 3, lcol = (tid & 7) * 8;  // loader coords (32x64/8)
  __bf16* Pw = &Ps[wid * 512];

  for (int kt = 0; kt < Nseq / 32; ++kt) {
    __syncthreads();
    {  // K tile row-major via async-to-LDS; V tile dh-transposed via regs
      async_copy_b128(&Ks[lrow * 64 + lcol],
                      kb + (size_t)(kt * 32 + lrow) * dhd + lcol);
      alignas(16) __bf16 tmp[8];
      *(uint4*)tmp = *(const uint4*)(vb + (size_t)(kt * 32 + lrow) * dhd + lcol);
#pragma unroll
      for (int i = 0; i < 8; ++i) Vst[(lcol + i) * 32 + lrow] = tmp[i];
    }
    wait_async0();
    __syncthreads();

    // S = Q @ K^T : two 16x16 tiles over 32 keys, K-dim = dh = 2x32
    v8f s[2];
#pragma unroll
    for (int nt = 0; nt < 2; ++nt) {
      v8f sc = zero;
#pragma unroll
      for (int c = 0; c < 2; ++c) {
        const __bf16* p = &Ks[(nt * 16 + (lane & 15)) * 64 + c * 32 + (lane >> 4) * 16];
        v16bf kf = mk16(*(const v8bf*)p, *(const v8bf*)(p + 8));
        sc = __builtin_amdgcn_wmma_f32_16x16x32_bf16(false, qf[c], false, kf,
                                                     (short)0, sc, false, false);
      }
      s[nt] = sc;
    }

    // Online softmax per row (row stats live on all 16 lanes of a half-group)
    float p0[8], p1[8], fac[8];
#pragma unroll
    for (int r = 0; r < 8; ++r) {
      float s0 = s[0][r] * 0.125f, s1 = s[1][r] * 0.125f;  // SCALE = 64^-0.5
      float mx = fmaxf(s0, s1);
      mx = fmaxf(mx, __shfl_xor(mx, 1));
      mx = fmaxf(mx, __shfl_xor(mx, 2));
      mx = fmaxf(mx, __shfl_xor(mx, 4));
      mx = fmaxf(mx, __shfl_xor(mx, 8));
      float m_new = fmaxf(m_run[r], mx);
      fac[r] = expf(m_run[r] - m_new);
      p0[r] = expf(s0 - m_new);
      p1[r] = expf(s1 - m_new);
      float rs = p0[r] + p1[r];
      rs += __shfl_xor(rs, 1);
      rs += __shfl_xor(rs, 2);
      rs += __shfl_xor(rs, 4);
      rs += __shfl_xor(rs, 8);
      l_run[r] = l_run[r] * fac[r] + rs;
      m_run[r] = m_new;
    }
#pragma unroll
    for (int o = 0; o < 4; ++o)
#pragma unroll
      for (int r = 0; r < 8; ++r) acc[o][r] *= fac[r];

    // Stage P (C-layout) into LDS, reload in A-layout (same-wave DS ordering)
#pragma unroll
    for (int r = 0; r < 8; ++r) {
      int rr = (lane >> 4) * 8 + r;
      Pw[rr * 32 + (lane & 15)]      = f2bf(p0[r]);
      Pw[rr * 32 + 16 + (lane & 15)] = f2bf(p1[r]);
    }
    const __bf16* pp = &Pw[(lane & 15) * 32 + (lane >> 4) * 16];
    v16bf pf = mk16(*(const v8bf*)pp, *(const v8bf*)(pp + 8));

    // O += P @ V : 4 dh tiles, B-frags contiguous from transposed V
#pragma unroll
    for (int o = 0; o < 4; ++o) {
      const __bf16* p = &Vst[(o * 16 + (lane & 15)) * 32 + (lane >> 4) * 16];
      v16bf vf = mk16(*(const v8bf*)p, *(const v8bf*)(p + 8));
      acc[o] = __builtin_amdgcn_wmma_f32_16x16x32_bf16(false, pf, false, vf,
                                                       (short)0, acc[o], false, false);
    }
  }

  // Normalize and write [B,N,D] f32
#pragma unroll
  for (int r = 0; r < 8; ++r) {
    float inv = 1.0f / l_run[r];
    int n = qr0 + (lane >> 4) * 8 + r;
#pragma unroll
    for (int o = 0; o < 4; ++o)
      out[((size_t)(b * Nseq + n)) * Dm + h * dhd + o * 16 + (lane & 15)] =
          acc[o][r] * inv;
  }
}

// ---------------------------------------------------------------- add + LN
DEV float block_sum(float v, float* red, int tid) {
#pragma unroll
  for (int off = 16; off > 0; off >>= 1) v += __shfl_xor(v, off);
  __syncthreads();
  if ((tid & 31) == 0) red[tid >> 5] = v;
  __syncthreads();
  float t = 0.f;
#pragma unroll
  for (int i = 0; i < 8; ++i) t += red[i];
  return t;
}

__global__ __launch_bounds__(256) void add_ln_kernel(
    const float* __restrict__ xin, const float* __restrict__ res,
    const float* __restrict__ g, const float* __restrict__ be,
    float* __restrict__ outF, __bf16* __restrict__ outB) {
  __shared__ float red[8];
  const int row = blockIdx.x, tid = threadIdx.x;
  float y[4];
  float s = 0.f;
#pragma unroll
  for (int i = 0; i < 4; ++i) {
    int c = tid + i * 256;
    y[i] = xin[(size_t)row * Dm + c] + res[(size_t)row * Dm + c];
    s += y[i];
  }
  float mu = block_sum(s, red, tid) * (1.0f / Dm);
  float v = 0.f;
#pragma unroll
  for (int i = 0; i < 4; ++i) { float d = y[i] - mu; v += d * d; }
  float rstd = rsqrtf(block_sum(v, red, tid) * (1.0f / Dm) + 1e-5f);
#pragma unroll
  for (int i = 0; i < 4; ++i) {
    int c = tid + i * 256;
    float o = (y[i] - mu) * rstd * g[c] + be[c];
    outF[(size_t)row * Dm + c] = o;
    if (outB) outB[(size_t)row * Dm + c] = f2bf(o);
  }
}

// ---------------------------------------------------------------- launcher
extern "C" void kernel_launch(void* const* d_in, const int* in_sizes, int n_in,
                              void* d_out, int out_size, void* d_ws, size_t ws_size,
                              hipStream_t stream) {
  (void)in_sizes; (void)n_in; (void)out_size; (void)ws_size;
  const float* x    = (const float*)d_in[0];
  const float* wqkv = (const float*)d_in[1];
  const float* bqkv = (const float*)d_in[2];
  const float* w1   = (const float*)d_in[3];
  const float* b1   = (const float*)d_in[4];
  const float* w2   = (const float*)d_in[5];
  const float* b2   = (const float*)d_in[6];
  const float* g1   = (const float*)d_in[7];
  const float* be1  = (const float*)d_in[8];
  const float* g2   = (const float*)d_in[9];
  const float* be2  = (const float*)d_in[10];

  char* ws = (char*)d_ws;
  size_t off = 0;
  auto alloc = [&](size_t bytes) {
    char* p = ws + off;
    off += (bytes + 255) & ~(size_t)255;
    return p;
  };
  __bf16* xb    = (__bf16*)alloc((size_t)BN * Dm * 2);
  __bf16* wqkvb = (__bf16*)alloc((size_t)Dm * 3 * Dm * 2);
  __bf16* w1b   = (__bf16*)alloc((size_t)Dm * Ff * 2);
  __bf16* w2b   = (__bf16*)alloc((size_t)Ff * Dm * 2);
  __bf16* qkvb  = (__bf16*)alloc(3 * QS * 2);
  float*  attn  = (float*) alloc((size_t)BN * Dm * 4);
  float*  x1    = (float*) alloc((size_t)BN * Dm * 4);
  __bf16* x1b   = (__bf16*)alloc((size_t)BN * Dm * 2);
  __bf16* hbuf  = (__bf16*)alloc((size_t)BN * Ff * 2);
  float*  ffn   = (float*) alloc((size_t)BN * Dm * 4);

  cvt_bf16_kernel<<<2048, 256, 0, stream>>>(x,    xb,    (size_t)BN * Dm);
  cvt_bf16_kernel<<<2048, 256, 0, stream>>>(wqkv, wqkvb, (size_t)Dm * 3 * Dm);
  cvt_bf16_kernel<<<2048, 256, 0, stream>>>(w1,   w1b,   (size_t)Dm * Ff);
  cvt_bf16_kernel<<<2048, 256, 0, stream>>>(w2,   w2b,   (size_t)Ff * Dm);

  // QKV projection: [8192,1024] @ [1024,3072] -> q/k/v [B,H,N,64] bf16
  gemm_bf16_kernel<0><<<dim3(3 * Dm / 128, BN / 128), 256, 0, stream>>>(
      xb, wqkvb, bqkv, Dm, 3 * Dm, qkvb, nullptr);
  // Attention
  attn_kernel<<<dim3(Nseq / 128, Bq * Hh), 256, 0, stream>>>(qkvb, attn);
  // x1 = LN(x + attn), keep f32 + bf16
  add_ln_kernel<<<BN, 256, 0, stream>>>(x, attn, g1, be1, x1, x1b);
  // FFN: gelu(x1 @ w1 + b1) -> hbuf bf16
  gemm_bf16_kernel<1><<<dim3(Ff / 128, BN / 128), 256, 0, stream>>>(
      x1b, w1b, b1, Dm, Ff, hbuf, nullptr);
  // hbuf @ w2 + b2 -> ffn f32
  gemm_bf16_kernel<2><<<dim3(Dm / 128, BN / 128), 256, 0, stream>>>(
      hbuf, w2b, b2, Ff, Dm, nullptr, ffn);
  // out = LN(x1 + ffn)
  add_ln_kernel<<<BN, 256, 0, stream>>>(x1, ffn, g2, be2, (float*)d_out, nullptr);
}
